// GraphDTI_39316130627944
// MI455X (gfx1250) — compile-verified, matrix operand order checked by reading, not run on hardware
//
#include <hip/hip_runtime.h>
#include <hip/hip_bf16.h>
#include <math.h>

// ---------------- types for WMMA (gfx1250 / CDNA5, wave32) ----------------
typedef __bf16 bf16_t;
typedef bf16_t v16bf __attribute__((ext_vector_type(16)));
typedef bf16_t v8bf  __attribute__((ext_vector_type(8)));
typedef bf16_t v2bf  __attribute__((ext_vector_type(2)));
typedef float  v8f   __attribute__((ext_vector_type(8)));
typedef int    v4i_t __attribute__((vector_size(16)));   // matches builtin param type

#define GAS __attribute__((address_space(1)))
#define LAS __attribute__((address_space(3)))

#if defined(__HIP_DEVICE_COMPILE__) && \
    __has_builtin(__builtin_amdgcn_global_load_async_to_lds_b128) && \
    __has_builtin(__builtin_amdgcn_s_wait_asynccnt)
#define HAVE_ASYNC 1
#else
#define HAVE_ASYNC 0
#endif

// ---------------- problem constants ----------------
#define BSZ   64
#define NPER  1500
#define DEGC  24
#define NN    (BSZ * NPER)        /* 96000 nodes  */
#define EE    (NN * DEGC)         /* 2304000 edges */
#define GDIM  512
#define HDIM  256
#define PROT  1280
#define DRUGD 1024

#define ASTR  36                  /* A LDS row stride (f32), 144B: 16B-aligned, bank-spread */
#define BSTR  40                  /* B LDS col stride (bf16), 80B: 16B-aligned, bank-spread */

// convert one A row run (f32 LDS) into a WMMA A-fragment (ISA 7.12.2 16-bit layout)
__device__ __forceinline__ v16bf frag_a_f32(const float* row, int kb) {
    const float4 a0 = *(const float4*)(row + kb);
    const float4 a1 = *(const float4*)(row + kb + 4);
    const float4 a2 = *(const float4*)(row + 16 + kb);
    const float4 a3 = *(const float4*)(row + 16 + kb + 4);
    v16bf o;
    o[0]  = (bf16_t)a0.x; o[1]  = (bf16_t)a0.y; o[2]  = (bf16_t)a0.z; o[3]  = (bf16_t)a0.w;
    o[4]  = (bf16_t)a1.x; o[5]  = (bf16_t)a1.y; o[6]  = (bf16_t)a1.z; o[7]  = (bf16_t)a1.w;
    o[8]  = (bf16_t)a2.x; o[9]  = (bf16_t)a2.y; o[10] = (bf16_t)a2.z; o[11] = (bf16_t)a2.w;
    o[12] = (bf16_t)a3.x; o[13] = (bf16_t)a3.y; o[14] = (bf16_t)a3.z; o[15] = (bf16_t)a3.w;
    return o;
}

// =====================================================================
// WMMA GEMM: C[M,N] = act(A[M,K] @ W[K,N] + bias)
// block = 256 thr = 8 waves; tile 64(M) x 64(N); K-step 32.
// wave (w>>2, w&3): rows {wm, wm+32}+lrow, cols wn; 2 accumulators,
// shared B fragment -> 2x v_wmma_f32_16x16x32_bf16 per K-step per wave.
// A tile staged f32 (async global->LDS when available), B staged bf16
// TRANSPOSED ([col][k]) so fragments are contiguous ds_load_b128s.
// Requires M%64==0, N%64==0, K%32==0 (true for every GEMM here).
// =====================================================================
__global__ __launch_bounds__(256)
void wmma_gemm_bf16(const float* __restrict__ A, const float* __restrict__ W,
                    const float* __restrict__ bias, float* __restrict__ C,
                    int M, int N, int K, int act)
{
    __shared__ float  Asf[64 * ASTR];   // 9216 B, f32 [row][k]
    __shared__ bf16_t Bst[64 * BSTR];   // 5120 B, bf16 [col][k] (transposed)

    const int tid  = threadIdx.x;
    const int m0   = blockIdx.x * 64;
    const int n0   = blockIdx.y * 64;
    const int wave = tid >> 5;
    const int lane = tid & 31;
    const int wn   = (wave & 3) << 4;
    const int wm   = (wave >> 2) << 4;
    const int lrow = lane & 15;
    const int kb   = (lane >> 4) << 3;   // lanes 16..31 take the K+8 half

    v8f acc0 = {0.f,0.f,0.f,0.f,0.f,0.f,0.f,0.f};
    v8f acc1 = {0.f,0.f,0.f,0.f,0.f,0.f,0.f,0.f};

    // B staging: thread -> k-pair (2 rows of W) x 4 cols
    const int bkp = tid >> 4;            // 0..15 -> k = 2*bkp, 2*bkp+1
    const int bcq = (tid & 15) * 4;      // col 0..60
#if HAVE_ASYNC
    const int ar0 = tid >> 3;            // rows 0..31
    const int af0 = (tid & 7) * 4;       // 16B chunk within row
#else
    const int arf = tid >> 2;            // rows 0..63, 8 f32 each
    const int afc = (tid * 8) & 31;
#endif

    for (int k0 = 0; k0 < K; k0 += 32) {
        // ---- stage A tile (64x32 f32) ----
#if HAVE_ASYNC
        __builtin_amdgcn_global_load_async_to_lds_b128(
            (GAS v4i_t*)(GAS void*)(uintptr_t)(A + (size_t)(m0 + ar0) * K + k0 + af0),
            (LAS v4i_t*)(LAS void*)&Asf[ar0 * ASTR + af0], 0, 0);
        __builtin_amdgcn_global_load_async_to_lds_b128(
            (GAS v4i_t*)(GAS void*)(uintptr_t)(A + (size_t)(m0 + ar0 + 32) * K + k0 + af0),
            (LAS v4i_t*)(LAS void*)&Asf[(ar0 + 32) * ASTR + af0], 0, 0);
#else
        {
            const float4 v0 = *(const float4*)(A + (size_t)(m0 + arf) * K + k0 + afc);
            const float4 v1 = *(const float4*)(A + (size_t)(m0 + arf) * K + k0 + afc + 4);
            *(float4*)&Asf[arf * ASTR + afc]     = v0;
            *(float4*)&Asf[arf * ASTR + afc + 4] = v1;
        }
#endif
        // ---- stage B tile (32x64 f32 -> bf16 transposed) ----
        {
            const float4 r0 = *(const float4*)(W + (size_t)(k0 + 2 * bkp) * N + n0 + bcq);
            const float4 r1 = *(const float4*)(W + (size_t)(k0 + 2 * bkp + 1) * N + n0 + bcq);
            const float r0a[4] = {r0.x, r0.y, r0.z, r0.w};
            const float r1a[4] = {r1.x, r1.y, r1.z, r1.w};
#pragma unroll
            for (int j = 0; j < 4; ++j) {
                v2bf p = {(bf16_t)r0a[j], (bf16_t)r1a[j]};
                *(v2bf*)&Bst[(bcq + j) * BSTR + 2 * bkp] = p;
            }
        }
        if (k0 + 32 < K)   // global_prefetch_b8 of next B slab
            __builtin_prefetch(W + (size_t)(k0 + 32 + 2 * bkp) * N + n0 + bcq, 0, 0);
#if HAVE_ASYNC
        __builtin_amdgcn_s_wait_asynccnt(0);
#endif
        __syncthreads();

        // ---- fragments ----
        const v16bf a0v = frag_a_f32(&Asf[(wm + lrow) * ASTR], kb);
        const v16bf a1v = frag_a_f32(&Asf[(wm + 32 + lrow) * ASTR], kb);
        const v8bf blo = *(const v8bf*)&Bst[(wn + lrow) * BSTR + kb];
        const v8bf bhi = *(const v8bf*)&Bst[(wn + lrow) * BSTR + 16 + kb];
        const v16bf bv = __builtin_shufflevector(blo, bhi,
            0,1,2,3,4,5,6,7,8,9,10,11,12,13,14,15);

        acc0 = __builtin_amdgcn_wmma_f32_16x16x32_bf16(false, a0v, false, bv,
                                                       (short)0, acc0, false, false);
        acc1 = __builtin_amdgcn_wmma_f32_16x16x32_bf16(false, a1v, false, bv,
                                                       (short)0, acc1, false, false);
        __syncthreads();
    }

    // ---- store: C/D layout — lane&15 = N col, lane>=16 adds 8 to M, vgpr r = M&7
    const int col    = n0 + wn + lrow;
    const int mbase0 = m0 + wm + ((lane >> 4) << 3);
    const float bv_  = bias ? bias[col] : 0.0f;
#pragma unroll
    for (int r = 0; r < 8; ++r) {
        float v = acc0[r] + bv_;
        if (act) v = fmaxf(v, 0.0f);
        C[(size_t)(mbase0 + r) * N + col] = v;
    }
#pragma unroll
    for (int r = 0; r < 8; ++r) {
        float v = acc1[r] + bv_;
        if (act) v = fmaxf(v, 0.0f);
        C[(size_t)(mbase0 + 32 + r) * N + col] = v;
    }
}

// ===================== GCN helper kernels =====================
__global__ void fill1_k(float* p, int n) {
    int i = blockIdx.x * blockDim.x + threadIdx.x;
    if (i < n) p[i] = 1.0f;
}
__global__ void deg_count_k(const int* __restrict__ dst, float* deg, int e) {
    int i = blockIdx.x * blockDim.x + threadIdx.x;
    if (i < e) atomicAdd(&deg[dst[i]], 1.0f);
}
__global__ void rsqrt_ip_k(float* p, int n) {
    int i = blockIdx.x * blockDim.x + threadIdx.x;
    if (i < n) p[i] = rsqrtf(p[i]);
}
// agg[i,f] = h[i,f]*dinv[i]^2 + bias[f]   (self-loop term, initializes agg)
__global__ void gcn_self_k(const float* __restrict__ h, const float* __restrict__ dinv,
                           const float* __restrict__ bias, float* __restrict__ agg,
                           int n, int F) {
    size_t i = (size_t)blockIdx.x * blockDim.x + threadIdx.x;
    if (i >= (size_t)n * F) return;
    int row = (int)(i / F), f = (int)(i % F);
    float d = dinv[row];
    agg[i] = h[i] * d * d + bias[f];
}
// one block per edge; blockDim = F/4; agg[dst] += h[src] * dinv[src]*dinv[dst]
__global__ void gcn_edge_k(const float* __restrict__ h, const float* __restrict__ dinv,
                           const int* __restrict__ src, const int* __restrict__ dst,
                           float* __restrict__ agg, int F) {
    int ed = blockIdx.x;
    int f4 = threadIdx.x * 4;
    int s = src[ed], d = dst[ed];
    float w = dinv[s] * dinv[d];
    float4 hv = *(const float4*)(h + (size_t)s * F + f4);
    float* ap = agg + (size_t)d * F + f4;
    atomicAdd(ap + 0, hv.x * w);
    atomicAdd(ap + 1, hv.y * w);
    atomicAdd(ap + 2, hv.z * w);
    atomicAdd(ap + 3, hv.w * w);
}
__global__ void relu_ip_k(float* x, size_t n) {
    size_t i = (size_t)blockIdx.x * blockDim.x + threadIdx.x;
    if (i < n) x[i] = fmaxf(x[i], 0.0f);
}
// io[i] = a[i] + relu(io[i])
__global__ void add_relu_k(const float* __restrict__ a, float* io, size_t n) {
    size_t i = (size_t)blockIdx.x * blockDim.x + threadIdx.x;
    if (i < n) io[i] = a[i] + fmaxf(io[i], 0.0f);
}

// ===================== attention head kernels =====================
__global__ void qproj_k(const float* __restrict__ S, const float* __restrict__ Wq,
                        const float* __restrict__ bq, float* q) {
    int j = threadIdx.x;
    float s = bq[j];
    for (int i = 0; i < HDIM; ++i) s += S[i] * Wq[i * HDIM + j];
    q[j] = s;
}
// grid (B, NHEADS); softmax over NPER keys; o = Qh + A@Vh
__global__ __launch_bounds__(256)
void attn_pool_k(const float* __restrict__ Kf, const float* __restrict__ Vf,
                 const float* __restrict__ q, float* __restrict__ o) {
    __shared__ float s[NPER];
    __shared__ float red[256];
    __shared__ float qh[32];
    const int b = blockIdx.x, h = blockIdx.y, tid = threadIdx.x;
    if (tid < 32) qh[tid] = q[h * 32 + tid];
    __syncthreads();
    float lmax = -1e30f;
    for (int k = tid; k < NPER; k += 256) {
        const float* kp = Kf + ((size_t)(b * NPER + k)) * HDIM + h * 32;
        float sc = 0.f;
#pragma unroll
        for (int j = 0; j < 32; ++j) sc += qh[j] * kp[j];
        sc *= 0.0625f;               // 1/sqrt(256)
        s[k] = sc;
        lmax = fmaxf(lmax, sc);
    }
    red[tid] = lmax; __syncthreads();
    for (int st = 128; st > 0; st >>= 1) { if (tid < st) red[tid] = fmaxf(red[tid], red[tid + st]); __syncthreads(); }
    float mx = red[0]; __syncthreads();
    float lsum = 0.f;
    for (int k = tid; k < NPER; k += 256) { float e = __expf(s[k] - mx); s[k] = e; lsum += e; }
    red[tid] = lsum; __syncthreads();
    for (int st = 128; st > 0; st >>= 1) { if (tid < st) red[tid] += red[tid + st]; __syncthreads(); }
    float inv = 1.0f / red[0];
    __syncthreads();
    if (tid < 32) {
        float accv = 0.f;
        for (int k = 0; k < NPER; ++k)
            accv += s[k] * Vf[((size_t)(b * NPER + k)) * HDIM + h * 32 + tid];
        o[b * HDIM + h * 32 + tid] = qh[tid] + accv * inv;
    }
}
// per-row layernorm, blockDim == 256 == HDIM, grid = rows
__global__ __launch_bounds__(256)
void layernorm_k(float* __restrict__ x, const float* __restrict__ g,
                 const float* __restrict__ b) {
    __shared__ float red[256];
    const int row = blockIdx.x, j = threadIdx.x;
    float v = x[row * HDIM + j];
    red[j] = v; __syncthreads();
    for (int st = 128; st > 0; st >>= 1) { if (j < st) red[j] += red[j + st]; __syncthreads(); }
    float mean = red[0] * (1.0f / HDIM); __syncthreads();
    float dv = v - mean;
    red[j] = dv * dv; __syncthreads();
    for (int st = 128; st > 0; st >>= 1) { if (j < st) red[j] += red[j + st]; __syncthreads(); }
    float var = red[0] * (1.0f / HDIM);
    x[row * HDIM + j] = dv * rsqrtf(var + 1e-5f) * g[j] + b[j];
}
__global__ void concat2_k(const float* __restrict__ a, const float* __restrict__ b2,
                          float* __restrict__ o) {
    int i = blockIdx.x * blockDim.x + threadIdx.x;   // BSZ*512 total
    int r = i >> 9, c = i & 511;
    o[i] = (c < HDIM) ? a[r * HDIM + c] : b2[r * HDIM + c - HDIM];
}
__global__ __launch_bounds__(256)
void out_proj_k(const float* __restrict__ x3, const float* __restrict__ w,
                const float* __restrict__ bout, float* __restrict__ out) {
    __shared__ float red[256];
    const int b = blockIdx.x, j = threadIdx.x;
    red[j] = x3[b * HDIM + j] * w[j];
    __syncthreads();
    for (int st = 128; st > 0; st >>= 1) { if (j < st) red[j] += red[j + st]; __syncthreads(); }
    if (j == 0) out[b] = red[0] + bout[0];
}

// =====================================================================
extern "C" void kernel_launch(void* const* d_in, const int* in_sizes, int n_in,
                              void* d_out, int out_size, void* d_ws, size_t ws_size,
                              hipStream_t stream) {
    (void)in_sizes; (void)n_in; (void)out_size; (void)ws_size;

    const float* drug   = (const float*)d_in[0];
    const float* prot_x = (const float*)d_in[1];
    const int*   esrc   = (const int*)d_in[2];
    const int*   edst   = (const int*)d_in[3];
    const float* esm_W  = (const float*)d_in[4];  const float* esm_b = (const float*)d_in[5];
    const float* g0_W   = (const float*)d_in[6];  const float* g0_b  = (const float*)d_in[7];
    const float* g1_W   = (const float*)d_in[8];  const float* g1_b  = (const float*)d_in[9];
    const float* g2_W   = (const float*)d_in[10]; const float* g2_b  = (const float*)d_in[11];
    const float* lin1_W = (const float*)d_in[12]; const float* lin1_b= (const float*)d_in[13];
    const float* mt_S   = (const float*)d_in[14];
    const float* q_W    = (const float*)d_in[15]; const float* q_b   = (const float*)d_in[16];
    const float* k_W    = (const float*)d_in[17]; const float* k_b   = (const float*)d_in[18];
    const float* v_W    = (const float*)d_in[19]; const float* v_b   = (const float*)d_in[20];
    const float* o_W    = (const float*)d_in[21]; const float* o_b   = (const float*)d_in[22];
    const float* lin2_W = (const float*)d_in[23]; const float* lin2_b= (const float*)d_in[24];
    const float* dl_W   = (const float*)d_in[25]; const float* dl_b  = (const float*)d_in[26];
    const float* h2_W   = (const float*)d_in[27]; const float* h2_b  = (const float*)d_in[28];
    const float* dp_W   = (const float*)d_in[29]; const float* dp_b  = (const float*)d_in[30];
    const float* out_W  = (const float*)d_in[31]; const float* out_b = (const float*)d_in[32];
    const float* ln0_g  = (const float*)d_in[33]; const float* ln0_b = (const float*)d_in[34];
    const float* ln1_g  = (const float*)d_in[35]; const float* ln1_b = (const float*)d_in[36];

    // ---------------- workspace carving (floats) ----------------
    float* b0   = (float*)d_ws;
    float* b1   = b0 + (size_t)NN * GDIM;
    float* b2   = b1 + (size_t)NN * GDIM;
    float* dinv = b2 + (size_t)NN * GDIM;
    float* qv   = dinv + NN;
    float* obuf = qv + HDIM;
    float* tmp  = obuf + BSZ * HDIM;
    float* dbuf = tmp + BSZ * HDIM;
    float* catb = dbuf + BSZ * 1024;
    float* x3   = catb + BSZ * 2 * HDIM;
    float* Kagg = b0 + (size_t)NN * HDIM;   // second half of b0
    float* Vagg = b2 + (size_t)NN * HDIM;   // second half of b2

    float* outp  = (float*)d_out;           // (64,)
    float* x2    = outp + BSZ;              // (64,256)
    float* gfeat = x2 + BSZ * HDIM;         // (64,256)

#define GEMM(Ap, Wp, Bp, Cp, M_, N_, K_, ACT) \
    wmma_gemm_bf16<<<dim3((M_)/64, (N_)/64), 256, 0, stream>>>(Ap, Wp, Bp, Cp, M_, N_, K_, ACT)

    const size_t NG = (size_t)NN * GDIM;
    const size_t NH = (size_t)NN * HDIM;
    const int EW_BLKS = (int)(NG / 256);
    const int EH_BLKS = (int)(NH / 256);

    // ---- degree -> dinv ----
    fill1_k<<<(NN + 255) / 256, 256, 0, stream>>>(dinv, NN);
    deg_count_k<<<(EE + 255) / 256, 256, 0, stream>>>(edst, dinv, EE);
    rsqrt_ip_k<<<(NN + 255) / 256, 256, 0, stream>>>(dinv, NN);

    // ---- protein branch ----
    GEMM(prot_x, esm_W, esm_b, b0, NN, GDIM, PROT, 0);                   // x
    GEMM(b0, g0_W, nullptr, b1, NN, GDIM, GDIM, 0);                      // lin0
    gcn_self_k<<<EW_BLKS, 256, 0, stream>>>(b1, dinv, g0_b, b2, NN, GDIM);
    gcn_edge_k<<<EE, GDIM / 4, 0, stream>>>(b1, dinv, esrc, edst, b2, GDIM);
    relu_ip_k<<<EW_BLKS, 256, 0, stream>>>(b2, NG);                      // h1 in b2
    GEMM(b2, g1_W, nullptr, b1, NN, GDIM, GDIM, 0);
    gcn_self_k<<<EW_BLKS, 256, 0, stream>>>(b1, dinv, g1_b, b0, NN, GDIM);
    gcn_edge_k<<<EE, GDIM / 4, 0, stream>>>(b1, dinv, esrc, edst, b0, GDIM);
    add_relu_k<<<EW_BLKS, 256, 0, stream>>>(b2, b0, NG);                 // h2 in b0
    GEMM(b0, g2_W, nullptr, b1, NN, GDIM, GDIM, 0);
    gcn_self_k<<<EW_BLKS, 256, 0, stream>>>(b1, dinv, g2_b, b2, NN, GDIM);
    gcn_edge_k<<<EE, GDIM / 4, 0, stream>>>(b1, dinv, esrc, edst, b2, GDIM);
    add_relu_k<<<EW_BLKS, 256, 0, stream>>>(b0, b2, NG);                 // h3 in b2

    // ---- GMT final PMA stage ----
    GEMM(b2, lin1_W, lin1_b, b1, NN, HDIM, GDIM, 0);                     // hh in b1
    GEMM(b1, k_W, nullptr, b0, NN, HDIM, HDIM, 0);                       // klin
    gcn_self_k<<<EH_BLKS, 256, 0, stream>>>(b0, dinv, k_b, Kagg, NN, HDIM);
    gcn_edge_k<<<EE, HDIM / 4, 0, stream>>>(b0, dinv, esrc, edst, Kagg, HDIM);
    GEMM(b1, v_W, nullptr, b2, NN, HDIM, HDIM, 0);                       // vlin
    gcn_self_k<<<EH_BLKS, 256, 0, stream>>>(b2, dinv, v_b, Vagg, NN, HDIM);
    gcn_edge_k<<<EE, HDIM / 4, 0, stream>>>(b2, dinv, esrc, edst, Vagg, HDIM);
    qproj_k<<<1, 256, 0, stream>>>(mt_S, q_W, q_b, qv);
    attn_pool_k<<<dim3(BSZ, 8), 256, 0, stream>>>(Kagg, Vagg, qv, obuf);
    layernorm_k<<<BSZ, 256, 0, stream>>>(obuf, ln0_g, ln0_b);
    GEMM(obuf, o_W, o_b, tmp, BSZ, HDIM, HDIM, 0);
    add_relu_k<<<(BSZ * HDIM) / 256, 256, 0, stream>>>(obuf, tmp, (size_t)BSZ * HDIM); // o in tmp
    layernorm_k<<<BSZ, 256, 0, stream>>>(tmp, ln1_g, ln1_b);
    GEMM(tmp, lin2_W, lin2_b, gfeat, BSZ, HDIM, HDIM, 0);                // g_feat -> d_out

    // ---- drug branch + fusion head ----
    GEMM(drug, dl_W, dl_b, dbuf, BSZ, 1024, DRUGD, 1);                   // relu(drug@dl)
    GEMM(dbuf, h2_W, h2_b, x2, BSZ, HDIM, 1024, 0);                      // x2 -> d_out
    concat2_k<<<(BSZ * 2 * HDIM) / 256, 256, 0, stream>>>(x2, gfeat, catb);
    GEMM(catb, dp_W, dp_b, x3, BSZ, HDIM, 2 * HDIM, 1);                  // relu(...)
    out_proj_k<<<BSZ, 256, 0, stream>>>(x3, out_W, out_b, outp);         // out -> d_out
#undef GEMM
}
